// GATBlock_87196426043536
// MI455X (gfx1250) — matrix operand order repaired and checked by reading, not compiled, not run on hardware
//
#include <hip/hip_runtime.h>
#include <hip/hip_bf16.h>

#define BB 16
#define NN 1024
#define EE 32768
#define HEADS 4
#define OC 8
#define D_IN 33
#define A_DIM 8
#define DMM 16
#define DOO 32
#define NODE_F 42                 // D_IN + A_DIM + 1
#define OUT0 (BB*NN*DOO)          // 524288 floats, then A follows in d_out

typedef __attribute__((ext_vector_type(16))) __bf16 v16bf;
typedef __attribute__((ext_vector_type(8)))  float  v8f;

static __device__ __forceinline__ v8f wmma_bf16(v16bf a, v16bf b, v8f c) {
  // D = A(16x32 bf16) * B(32x16 bf16) + C(16x16 f32)
  return __builtin_amdgcn_wmma_f32_16x16x32_bf16(false, a, false, b, (short)0, c, false, false);
}

// A fragment (16x32, bf16) gathered from f32 LDS tile src[16][ld].
// Layout (ISA 7.12.2): lane<16 -> M=lane, K elems {0..7,16..23}; lane>=16 -> M=lane-16, K {8..15,24..31}.
static __device__ __forceinline__ v16bf afragf(const float* src, int ld, int kOff, int kvalid, int lane) {
  int m   = lane & 15;
  int kb0 = (lane & 16) ? 8 : 0;
  v16bf a;
#pragma unroll
  for (int i = 0; i < 8; ++i) {
    int k0 = kb0 + i;
    int k1 = kb0 + 16 + i;
    a[i]     = (k0 < kvalid) ? (__bf16)src[m*ld + kOff + k0] : (__bf16)0.f;
    a[i + 8] = (k1 < kvalid) ? (__bf16)src[m*ld + kOff + k1] : (__bf16)0.f;
  }
  return a;
}

// B fragment (32x16, bf16): B[k][n] = W[n*sn + (kOff+k)*sk], n = nBase + lane%16.
// lane<16 holds rows k={0..7,16..23} of its column; lane>=16 holds k={8..15,24..31}.
static __device__ __forceinline__ v16bf bfragf(const float* W, int sn, int sk, int nBase, int kOff, int kvalid, int lane) {
  int n  = nBase + (lane & 15);
  int kb = (lane & 16) ? 8 : 0;
  v16bf b;
#pragma unroll
  for (int i = 0; i < 16; ++i) {
    int k = kb + (i < 8 ? i : i + 8);
    b[i] = (k < kvalid) ? (__bf16)W[n*sn + (kOff + k)*sk] : (__bf16)0.f;
  }
  return b;
}

// Scatter 16x16 f32 C/D accumulator into LDS dst[16][ld] at column base nBase.
// Layout: lane<16 -> N=lane, M=r; lane>=16 -> N=lane-16, M=r+8.
static __device__ __forceinline__ void storeD(float* dst, int ld, int nBase, v8f d, int lane) {
  int n  = nBase + (lane & 15);
  int mb = (lane & 16) ? 8 : 0;
#pragma unroll
  for (int r = 0; r < 8; ++r) dst[(mb + r)*ld + n] = d[r];
}

#define WAVES_PER_BLOCK 2
#define ARENA_F 3328   // floats per wave arena (overlaid phase buffers)

__global__ __launch_bounds__(64) void node_kernel(
    const float* __restrict__ H,
    const float* __restrict__ fc_s_w, const float* __restrict__ fc_s_b,
    const float* __restrict__ fc_a_w, const float* __restrict__ fc_a_b,
    const float* __restrict__ fc_r_w, const float* __restrict__ fc_r_b,
    const float* __restrict__ fuse_q,
    const float* __restrict__ fuse_k_w, const float* __restrict__ fuse_v_w,
    const float* __restrict__ proj_w, const float* __restrict__ proj_b,
    const float* __restrict__ post_w, const float* __restrict__ post_b,
    const float* __restrict__ ln_g, const float* __restrict__ ln_b,
    const float* __restrict__ gat_w,
    const float* __restrict__ att_src, const float* __restrict__ att_dst,
    float* __restrict__ xbuf, float* __restrict__ asrc, float* __restrict__ adst)
{
  __shared__ float smem[6432 + WAVES_PER_BLOCK * ARENA_F];
  float* wtokB = smem;          // [64][48] block-diagonal combined tok weight, k-major
  float* tokb  = wtokB + 3072;  // 48
  float* fq    = tokb + 48;     // 48
  float* fkw   = fq + 48;       // 256
  float* fvw   = fkw + 256;     // 256
  float* prw   = fvw + 256;     // 512
  float* prb   = prw + 512;     // 32
  float* pstw  = prb + 32;      // 1024
  float* pstb  = pstw + 1024;   // 32
  float* lng   = pstb + 32;     // 32
  float* lnb   = lng + 32;      // 32
  float* gw    = lnb + 32;      // 1024
  float* ats   = gw + 1024;     // 32
  float* atd   = ats + 32;      // 32
  float* arenas = atd + 32;

  const int tid = threadIdx.x, nth = blockDim.x;

  // Build combined tok weight W^T (K=42 padded to 64, N=48): cols 0..32 -> fc_s, 33..40 -> fc_a, 41 -> fc_r
  for (int i = tid; i < 3072; i += nth) {
    int k = i / 48, n = i % 48;
    float v = 0.f;
    if (n < 16)      { if (k < D_IN)                      v = fc_s_w[n*D_IN + k]; }
    else if (n < 32) { if (k >= D_IN && k < D_IN + A_DIM) v = fc_a_w[(n-16)*A_DIM + (k-D_IN)]; }
    else             { if (k == D_IN + A_DIM)             v = fc_r_w[n-32]; }
    wtokB[i] = v;
  }
  for (int i = tid; i < 48;   i += nth) tokb[i] = (i < 16) ? fc_s_b[i] : (i < 32) ? fc_a_b[i-16] : fc_r_b[i-32];
  for (int i = tid; i < 48;   i += nth) fq[i] = fuse_q[i];
  for (int i = tid; i < 256;  i += nth) { fkw[i] = fuse_k_w[i]; fvw[i] = fuse_v_w[i]; }
  for (int i = tid; i < 512;  i += nth) prw[i] = proj_w[i];
  for (int i = tid; i < 1024; i += nth) { pstw[i] = post_w[i]; gw[i] = gat_w[i]; }
  for (int i = tid; i < 32;   i += nth) {
    prb[i] = proj_b[i]; pstb[i] = post_b[i]; lng[i] = ln_g[i]; lnb[i] = ln_b[i];
    ats[i] = att_src[i]; atd[i] = att_dst[i];
  }
  __syncthreads();

  const int wave = tid >> 5, lane = tid & 31;
  float* AR       = arenas + wave * ARENA_F;
  float* hA       = AR;           // [16][64]   (phase 1)
  float* tokLds   = AR + 1024;    // [16][48]
  float* kLds     = AR + 1792;    // [16][48]
  float* vLds     = AR + 2560;    // [16][48]
  float* fusedLds = AR;           // [16][16]   (reuse hA)
  float* zLds     = AR + 256;     // [16][32]
  float* hLds     = AR + 768;     // [16][32]
  float* HnLds    = AR + 1280;    // [16][32]
  float* xLds     = AR + 1792;    // [16][32]   (reuse kLds)

  const int gnode0 = (blockIdx.x * WAVES_PER_BLOCK + wave) * 16;  // flat b*N+n tile base

  // Stage 16 contiguous H rows, zero-padded to 64 cols
  {
    const float* hsrc = H + (size_t)gnode0 * NODE_F;
    for (int i = lane; i < 16*64; i += 32) {
      int m = i >> 6, c = i & 63;
      hA[i] = (c < NODE_F) ? hsrc[m*NODE_F + c] : 0.f;
    }
  }
  __syncthreads();

  // tok (16x48) = Hrows(16x42) @ Wtok^T : 2 k-steps x 3 n-tiles of WMMA
  {
    v16bf a0 = afragf(hA, 64, 0,  32, lane);
    v16bf a1 = afragf(hA, 64, 32, 32, lane);
#pragma unroll
    for (int t = 0; t < 3; ++t) {
      v8f acc = {};
      acc = wmma_bf16(a0, bfragf(wtokB, 1, 48, t*16, 0,  32, lane), acc);
      acc = wmma_bf16(a1, bfragf(wtokB, 1, 48, t*16, 32, 32, lane), acc);
      float bb = tokb[t*16 + (lane & 15)];
#pragma unroll
      for (int r = 0; r < 8; ++r) acc[r] += bb;
      storeD(tokLds, 48, t*16, acc, lane);
    }
  }
  __syncthreads();

  // K_t / V_t = tok_t @ fuse_{k,v}_w^T (16x16, K=16)
#pragma unroll
  for (int t = 0; t < 3; ++t) {
    v16bf a = afragf(tokLds, 48, t*16, 16, lane);
    v8f kacc = {}, vacc = {};
    kacc = wmma_bf16(a, bfragf(fkw, 16, 1, 0, 0, 16, lane), kacc);
    vacc = wmma_bf16(a, bfragf(fvw, 16, 1, 0, 0, 16, lane), vacc);
    storeD(kLds, 48, t*16, kacc, lane);
    storeD(vLds, 48, t*16, vacc, lane);
  }
  __syncthreads();

  // Per-node 3x3 fusion attention: softmax over t, mean over q folds to per-t coefficients
  if (lane < 16) {
    int m = lane;
    float coef0 = 0.f, coef1 = 0.f, coef2 = 0.f;
#pragma unroll
    for (int q = 0; q < 3; ++q) {
      float at[3];
#pragma unroll
      for (int t = 0; t < 3; ++t) {
        float s = 0.f;
#pragma unroll
        for (int d = 0; d < 16; ++d) s += fq[q*16 + d] * kLds[m*48 + t*16 + d];
        at[t] = s * 0.25f;  // 1/sqrt(DM)
      }
      float mx = fmaxf(at[0], fmaxf(at[1], at[2]));
      float e0 = __expf(at[0]-mx), e1 = __expf(at[1]-mx), e2 = __expf(at[2]-mx);
      float inv = 1.f / (e0 + e1 + e2);
      coef0 += e0*inv; coef1 += e1*inv; coef2 += e2*inv;
    }
#pragma unroll
    for (int d = 0; d < 16; ++d) {
      float f = (coef0*vLds[m*48 + d] + coef1*vLds[m*48 + 16 + d] + coef2*vLds[m*48 + 32 + d]) * (1.f/3.f);
      fusedLds[m*16 + d] = fmaxf(f, 0.f);   // relu(fused)
    }
  }
  __syncthreads();

  // z = relu(fused) @ proj_w^T + proj_b : (16x16)@(16x32)
  {
    v16bf a = afragf(fusedLds, 16, 0, 16, lane);
#pragma unroll
    for (int t2 = 0; t2 < 2; ++t2) {
      v8f acc = {};
      acc = wmma_bf16(a, bfragf(prw, 16, 1, t2*16, 0, 16, lane), acc);
      float bb = prb[t2*16 + (lane & 15)];
#pragma unroll
      for (int r = 0; r < 8; ++r) acc[r] += bb;
      storeD(zLds, 32, t2*16, acc, lane);
    }
  }
  __syncthreads();

  // h = relu(z @ post_w^T + post_b) : (16x32)@(32x32), full K=32 in one WMMA
  {
    v16bf a = afragf(zLds, 32, 0, 32, lane);
#pragma unroll
    for (int t2 = 0; t2 < 2; ++t2) {
      v8f acc = {};
      acc = wmma_bf16(a, bfragf(pstw, 32, 1, t2*16, 0, 32, lane), acc);
      float bb = pstb[t2*16 + (lane & 15)];
#pragma unroll
      for (int r = 0; r < 8; ++r) acc[r] = fmaxf(acc[r] + bb, 0.f);
      storeD(hLds, 32, t2*16, acc, lane);
    }
  }
  __syncthreads();

  // LayerNorm over 32 features
  if (lane < 16) {
    int m = lane;
    float s = 0.f;
#pragma unroll
    for (int d = 0; d < 32; ++d) s += hLds[m*32 + d];
    float mu = s * (1.f/32.f);
    float v = 0.f;
#pragma unroll
    for (int d = 0; d < 32; ++d) { float t = hLds[m*32 + d] - mu; v += t*t; }
    float rinv = rsqrtf(v * (1.f/32.f) + 1e-5f);
#pragma unroll
    for (int d = 0; d < 32; ++d) HnLds[m*32 + d] = (hLds[m*32 + d] - mu) * rinv * lng[d] + lnb[d];
  }
  __syncthreads();

  // x = Hn @ gat_w^T : (16x32)@(32x32)
  {
    v16bf a = afragf(HnLds, 32, 0, 32, lane);
#pragma unroll
    for (int t2 = 0; t2 < 2; ++t2) {
      v8f acc = {};
      acc = wmma_bf16(a, bfragf(gw, 32, 1, t2*16, 0, 32, lane), acc);
      storeD(xLds, 32, t2*16, acc, lane);
    }
  }
  __syncthreads();

  // Coalesced x write + per-head GAT logits (lanes<16 -> a_src, lanes>=16 -> a_dst)
  {
    size_t xb = (size_t)gnode0 * DOO;
    for (int i = lane; i < 16*DOO; i += 32) xbuf[xb + i] = xLds[i];
    int m = lane & 15;
    const float* att = (lane < 16) ? ats : atd;
    float* dbuf      = (lane < 16) ? asrc : adst;
#pragma unroll
    for (int h = 0; h < HEADS; ++h) {
      float s = 0.f;
#pragma unroll
      for (int oc = 0; oc < OC; ++oc) s += xLds[m*DOO + h*OC + oc] * att[h*OC + oc];
      dbuf[(size_t)(gnode0 + m)*HEADS + h] = s;
    }
  }
}

// ---------------- edge phase ----------------

static __device__ __forceinline__ float lrelu(float v) { return v > 0.f ? v : 0.2f * v; }

// order-preserving float<->uint encoding for atomicMax
static __device__ __forceinline__ unsigned fenc(float f) {
  int b = __float_as_int(f);
  return (b >= 0) ? ((unsigned)b | 0x80000000u) : ~(unsigned)b;
}
static __device__ __forceinline__ float fdec(unsigned u) {
  int b = (u & 0x80000000u) ? (int)(u & 0x7FFFFFFFu) : (int)~u;
  return __int_as_float(b);
}

__global__ __launch_bounds__(256) void zeroA_kernel(float4* __restrict__ A4, int n4) {
  int stride = gridDim.x * blockDim.x;
  for (int i = blockIdx.x * blockDim.x + threadIdx.x; i < n4; i += stride)
    A4[i] = make_float4(0.f, 0.f, 0.f, 0.f);
}

__global__ __launch_bounds__(256) void init_small_kernel(unsigned* __restrict__ menc,
                                                         float* __restrict__ den,
                                                         float* __restrict__ rowsum,
                                                         float* __restrict__ out,
                                                         const float* __restrict__ gat_b) {
  int i = blockIdx.x * blockDim.x + threadIdx.x;   // exactly OUT0 threads
  if (i < BB*NN*HEADS) { menc[i] = 0x007FFFFFu /* enc(-inf) */; den[i] = 0.f; rowsum[i] = 0.f; }
  out[i] = gat_b[i & (DOO - 1)];
}

__global__ __launch_bounds__(256) void edge_max_kernel(const int* __restrict__ ei,
                                                       const float* __restrict__ asrc,
                                                       const float* __restrict__ adst,
                                                       unsigned* __restrict__ menc) {
  int idx = blockIdx.x * blockDim.x + threadIdx.x;   // b*E + e (exact)
  int b = idx >> 15, e = idx & (EE - 1);
  int s = ei[e], d = ei[EE + e];
  const float* ps = asrc + (size_t)(b*NN + s) * HEADS;
  const float* pd = adst + (size_t)(b*NN + d) * HEADS;
  unsigned* pm = menc + (size_t)(b*NN + d) * HEADS;
#pragma unroll
  for (int h = 0; h < HEADS; ++h) atomicMax(&pm[h], fenc(lrelu(ps[h] + pd[h])));
}

__global__ __launch_bounds__(256) void edge_sum_kernel(const int* __restrict__ ei,
                                                       const float* __restrict__ asrc,
                                                       const float* __restrict__ adst,
                                                       const unsigned* __restrict__ menc,
                                                       float* __restrict__ den) {
  int idx = blockIdx.x * blockDim.x + threadIdx.x;
  int b = idx >> 15, e = idx & (EE - 1);
  int s = ei[e], d = ei[EE + e];
  const float* ps = asrc + (size_t)(b*NN + s) * HEADS;
  const float* pd = adst + (size_t)(b*NN + d) * HEADS;
  const unsigned* pm = menc + (size_t)(b*NN + d) * HEADS;
  float* pden = den + (size_t)(b*NN + d) * HEADS;
#pragma unroll
  for (int h = 0; h < HEADS; ++h) {
    float ev = lrelu(ps[h] + pd[h]);
    atomicAdd(&pden[h], __expf(ev - fdec(pm[h])));
  }
}

__global__ __launch_bounds__(256) void edge_out_kernel(const int* __restrict__ ei,
                                                       const float* __restrict__ asrc,
                                                       const float* __restrict__ adst,
                                                       const unsigned* __restrict__ menc,
                                                       const float* __restrict__ den,
                                                       float* __restrict__ rowsum,
                                                       const float* __restrict__ xbuf,
                                                       float* __restrict__ out) {
  int idx = blockIdx.x * blockDim.x + threadIdx.x;
  int b = idx >> 15, e = idx & (EE - 1);
  int s = ei[e], d = ei[EE + e];
  const float* ps = asrc + (size_t)(b*NN + s) * HEADS;
  const float* pd = adst + (size_t)(b*NN + d) * HEADS;
  const unsigned* pm = menc + (size_t)(b*NN + d) * HEADS;
  const float* pden  = den  + (size_t)(b*NN + d) * HEADS;
  float* prs = rowsum + (size_t)(b*NN + s) * HEADS;
  const float* xs = xbuf + (size_t)(b*NN + s) * DOO;
  float* po = out + (size_t)(b*NN + d) * DOO;
#pragma unroll
  for (int h = 0; h < HEADS; ++h) {
    float ev = lrelu(ps[h] + pd[h]);
    float al = __expf(ev - fdec(pm[h])) / pden[h];
    atomicAdd(&prs[h], al);
#pragma unroll
    for (int oc = 0; oc < OC; ++oc) atomicAdd(&po[h*OC + oc], al * xs[h*OC + oc]);
  }
}

__global__ __launch_bounds__(256) void edge_A_kernel(const int* __restrict__ ei,
                                                     const float* __restrict__ asrc,
                                                     const float* __restrict__ adst,
                                                     const unsigned* __restrict__ menc,
                                                     const float* __restrict__ den,
                                                     const float* __restrict__ rowsum,
                                                     float* __restrict__ A) {
  int idx = blockIdx.x * blockDim.x + threadIdx.x;
  int b = idx >> 15, e = idx & (EE - 1);
  int s = ei[e], d = ei[EE + e];
  const float* ps = asrc + (size_t)(b*NN + s) * HEADS;
  const float* pd = adst + (size_t)(b*NN + d) * HEADS;
  const unsigned* pm = menc + (size_t)(b*NN + d) * HEADS;
  const float* pden  = den  + (size_t)(b*NN + d) * HEADS;
  const float* prs   = rowsum + (size_t)(b*NN + s) * HEADS;
#pragma unroll
  for (int h = 0; h < HEADS; ++h) {
    float ev = lrelu(ps[h] + pd[h]);
    float al = __expf(ev - fdec(pm[h])) / pden[h];
    size_t off = (((size_t)b * HEADS + h) * NN + s) * NN + d;
    atomicAdd(&A[off], al / fmaxf(prs[h], 1e-9f));   // row-normalized on the fly
  }
}

extern "C" void kernel_launch(void* const* d_in, const int* in_sizes, int n_in,
                              void* d_out, int out_size, void* d_ws, size_t ws_size,
                              hipStream_t stream) {
  const float* H        = (const float*)d_in[0];
  const int*   ei       = (const int*)  d_in[1];
  const float* fc_s_w   = (const float*)d_in[2];
  const float* fc_s_b   = (const float*)d_in[3];
  const float* fc_a_w   = (const float*)d_in[4];
  const float* fc_a_b   = (const float*)d_in[5];
  const float* fc_r_w   = (const float*)d_in[6];
  const float* fc_r_b   = (const float*)d_in[7];
  const float* fuse_q   = (const float*)d_in[8];
  const float* fuse_k_w = (const float*)d_in[9];
  const float* fuse_v_w = (const float*)d_in[10];
  const float* proj_w   = (const float*)d_in[11];
  const float* proj_b   = (const float*)d_in[12];
  const float* post_w   = (const float*)d_in[13];
  const float* post_b   = (const float*)d_in[14];
  const float* ln_g     = (const float*)d_in[15];
  const float* ln_b     = (const float*)d_in[16];
  const float* gat_w    = (const float*)d_in[17];
  const float* att_src  = (const float*)d_in[18];
  const float* att_dst  = (const float*)d_in[19];
  const float* gat_b    = (const float*)d_in[20];

  float* out = (float*)d_out;
  float* A   = out + (size_t)OUT0;

  float*    ws     = (float*)d_ws;
  float*    xbuf   = ws;                         // B*N*32 = 524288
  float*    asrc   = xbuf + (size_t)BB*NN*DOO;   // B*N*4
  float*    adst   = asrc + (size_t)BB*NN*HEADS; // B*N*4
  unsigned* menc   = (unsigned*)(adst + (size_t)BB*NN*HEADS);
  float*    den    = adst + 2*(size_t)BB*NN*HEADS;
  float*    rowsum = den  + (size_t)BB*NN*HEADS;

  // 1) WMMA node pipeline -> x, a_src, a_dst
  node_kernel<<<(BB*NN)/(16*WAVES_PER_BLOCK), 32*WAVES_PER_BLOCK, 0, stream>>>(
      H, fc_s_w, fc_s_b, fc_a_w, fc_a_b, fc_r_w, fc_r_b,
      fuse_q, fuse_k_w, fuse_v_w, proj_w, proj_b, post_w, post_b,
      ln_g, ln_b, gat_w, att_src, att_dst, xbuf, asrc, adst);

  // 2) single zeroing pass over the 256MB A output (the bandwidth floor)
  zeroA_kernel<<<4096, 256, 0, stream>>>((float4*)A, (int)(((size_t)BB*HEADS*NN*NN) / 4));

  // 3) init segment buffers; out initialized to gat_b broadcast
  init_small_kernel<<<OUT0/256, 256, 0, stream>>>(menc, den, rowsum, out, gat_b);

  // 4) segment softmax over dst + scatter (B*E = 524288 threads each)
  edge_max_kernel<<<(BB*EE)/256, 256, 0, stream>>>(ei, asrc, adst, menc);
  edge_sum_kernel<<<(BB*EE)/256, 256, 0, stream>>>(ei, asrc, adst, menc, den);
  edge_out_kernel<<<(BB*EE)/256, 256, 0, stream>>>(ei, asrc, adst, menc, den, rowsum, xbuf, out);
  edge_A_kernel  <<<(BB*EE)/256, 256, 0, stream>>>(ei, asrc, adst, menc, den, rowsum, A);
}
